// MaskMLABlock_31688268709943
// MI455X (gfx1250) — matrix operand order repaired and checked by reading, not compiled, not run on hardware
//
#include <hip/hip_runtime.h>

// ---------------- problem constants ----------------
#define BATCH   4
#define CCH     270
#define HH      66
#define WW      66
#define HWSZ    (HH*WW)          // 4356
#define NPATCH  4096
#define NREG    3
#define NTOK    (NPATCH + NREG)  // 4099
#define MROWS   (BATCH*NTOK)     // 16396
#define MROWSP  16400            // padded to multiple of 16
#define DIMT    2430             // token dim
#define DIMP    2432             // padded K (x32)
#define DDOWN   486              // DIM/5
#define DDP     512              // padded
#define QKVD    1458             // 3*DDOWN
#define HEADS   6
#define HD      81               // DDOWN/HEADS
#define H1D     607              // DIM/4
#define H1P     608              // padded
#define CHUNK   512              // NTOK // 8
#define SCALE_ATT 0.04969039949999533f   // (2430/6)^-0.5
#define NEG_SLOPE 0.01f
// weight N-dim pads (multiples of 64)
#define NP_DOWN 512
#define NP_QKV  1472
#define NP_UP   2432
#define NP_MLP1 640
#define NP_MLP2 2432

typedef __attribute__((ext_vector_type(16))) _Float16 v16h;
typedef __attribute__((ext_vector_type(8)))  float    v8f;

// ---------------- WMMA fragment helpers (wave32) ----------------
// A 16x32 f16 fragment: lanes 0-15 rows M=0..15 hold K 0..15,
// lanes 16-31 hold K 16..31 (matches B). Unconditional (padded).
__device__ __forceinline__ v16h frag_a_full(const _Float16* __restrict__ p, int hf) {
  v16h f;
  int kb = hf * 16;
#pragma unroll
  for (int i = 0; i < 16; ++i) f[i] = p[kb + i];
  return f;
}
// B 32x16 f16 fragment: lane holds one N-column; lane half selects K 0..15 / 16..31.
__device__ __forceinline__ v16h frag_b_full(const _Float16* __restrict__ p, int hf) {
  v16h f;
  int kb = hf * 16;
#pragma unroll
  for (int i = 0; i < 16; ++i) f[i] = p[kb + i];
  return f;
}
// bounded variants (edge tiles; guards constant-fold when Kd is compile-time)
__device__ __forceinline__ v16h frag_a(const _Float16* __restrict__ rowp,
                                       int k0, int Kd, int hf) {
  v16h f = {};
  int kb = k0 + hf * 16;
#pragma unroll
  for (int i = 0; i < 16; ++i) { int k = kb + i; if (k < Kd) f[i] = rowp[k]; }
  return f;
}
__device__ __forceinline__ v16h frag_b_contig(const _Float16* __restrict__ colp,
                                              int k0, int Kd, int hf) {
  v16h f = {};
  int kb = k0 + hf * 16;
#pragma unroll
  for (int i = 0; i < 16; ++i) { int k = kb + i; if (k < Kd) f[i] = colp[k]; }
  return f;
}
__device__ __forceinline__ v16h frag_b_strided(const _Float16* __restrict__ colp,
                                               long ld, int k0, int Kd, int hf) {
  v16h f = {};
  int kb = k0 + hf * 16;
#pragma unroll
  for (int i = 0; i < 16; ++i) { int k = kb + i; if (k < Kd) f[i] = colp[(size_t)k * ld]; }
  return f;
}
__device__ __forceinline__ v8f wmma16(v16h a, v16h b, v8f c) {
  return __builtin_amdgcn_wmma_f32_16x16x32_f16(false, a, false, b, (short)0, c, false, false);
}

// ---------------- TN GEMM (padded operands): Y = X(MxK) * W(NxK)^T -------------
// wave computes 16(M) x 64(N) with 4 accumulators; inner loop is branch-free.
// OUTM: 0 = write f16, 2 = accumulate into fp32 (residual stream)
template<int OUTM, bool RELU, bool BIAS>
__global__ void kGemm(const _Float16* __restrict__ A, long lda,
                      const _Float16* __restrict__ Bw, long ldb,
                      const float* __restrict__ bias,
                      _Float16* __restrict__ outH, float* __restrict__ outF, long ldc,
                      int M, int N, int Npad, int Kp, float alpha) {
  int wave = threadIdx.x >> 5;
  int lane = threadIdx.x & 31;
  int hf = lane >> 4, r = lane & 15;
  int tn0 = (blockIdx.x * 4 + wave) * 64;
  int tm  = blockIdx.y * 16;
  if (tn0 >= Npad) return;                       // wave-uniform: EXEC stays all-ones
  const _Float16* ap = A + (size_t)(tm + r) * lda;
  const _Float16* bp = Bw + (size_t)(tn0 + r) * ldb;
  v8f acc0 = {}, acc1 = {}, acc2 = {}, acc3 = {};
  for (int k0 = 0; k0 < Kp; k0 += 32) {
    v16h af = frag_a_full(ap + k0, hf);
    __builtin_prefetch(ap + k0 + 256, 0, 0);     // global_prefetch_b8
    v16h b0 = frag_b_full(bp + k0, hf);
    v16h b1 = frag_b_full(bp + (size_t)16 * ldb + k0, hf);
    v16h b2 = frag_b_full(bp + (size_t)32 * ldb + k0, hf);
    v16h b3 = frag_b_full(bp + (size_t)48 * ldb + k0, hf);
    acc0 = wmma16(af, b0, acc0);
    acc1 = wmma16(af, b1, acc1);
    acc2 = wmma16(af, b2, acc2);
    acc3 = wmma16(af, b3, acc3);
  }
  auto store_tile = [&](v8f acc, int tn) {
    int col = tn + r;
    if (col >= N) return;
#pragma unroll
    for (int v = 0; v < 8; ++v) {
      int row = tm + hf * 8 + v;
      if (row < M) {
        float val = acc[v] * alpha;
        if (BIAS) val += bias[col];
        if (RELU) val = fmaxf(val, 0.f);
        if (OUTM == 0)      outH[(size_t)row * ldc + col] = (_Float16)val;
        else                outF[(size_t)row * ldc + col] += val;
      }
    }
  };
  store_tile(acc0, tn0);
  store_tile(acc1, tn0 + 16);
  store_tile(acc2, tn0 + 32);
  store_tile(acc3, tn0 + 48);
}

// ---------------- attention: S = Q K^T * scale (per chunk, batched over B*H) ----
// FULL: compile-time cs = CHUNK -> all row/col guards fold away.
template<bool FULL>
__global__ void kAttnScores(const _Float16* __restrict__ qkv, float* __restrict__ S,
                            int st, int cs_rt) {
  const int cs = FULL ? CHUNK : cs_rt;
  int bz = blockIdx.z; int b = bz / HEADS, h = bz % HEADS;
  int lane = threadIdx.x & 31; int hf = lane >> 4, r = lane & 15;
  int tm = blockIdx.y * 16, tn = blockIdx.x * 16;
  const _Float16* Qb = qkv + ((size_t)(b * NTOK + st)) * QKVD + h * HD;
  const _Float16* Kb = qkv + ((size_t)(b * NTOK + st)) * QKVD + DDOWN + h * HD;
  int arow = tm + r, bcol = tn + r;
  v8f acc = {};
#pragma unroll
  for (int k0 = 0; k0 < HD; k0 += 32) {        // HD compile-time: tail folds per element
    v16h af = {}, bf = {};
    if (FULL || arow < cs) af = frag_a(Qb + (size_t)arow * QKVD, k0, HD, hf);
    if (FULL || bcol < cs) bf = frag_b_contig(Kb + (size_t)bcol * QKVD, k0, HD, hf);
    acc = wmma16(af, bf, acc);
  }
  float* Sp = S + (size_t)bz * CHUNK * CHUNK;
  if (FULL || bcol < cs) {
#pragma unroll
    for (int v = 0; v < 8; ++v) {
      int row = tm + hf * 8 + v;
      if (FULL || row < cs) Sp[(size_t)row * CHUNK + bcol] = acc[v] * SCALE_ATT;
    }
  }
}

// ---------------- attention: O = P V (per chunk) ----------------
template<bool FULL>
__global__ void kAttnAV(const _Float16* __restrict__ P, const _Float16* __restrict__ qkv,
                        _Float16* __restrict__ O, int st, int cs_rt) {
  const int cs = FULL ? CHUNK : cs_rt;
  int bz = blockIdx.z; int b = bz / HEADS, h = bz % HEADS;
  int lane = threadIdx.x & 31; int hf = lane >> 4, r = lane & 15;
  int tm = blockIdx.y * 16, tn = blockIdx.x * 16;
  const _Float16* Pp = P + (size_t)bz * CHUNK * CHUNK;
  const _Float16* Vb = qkv + ((size_t)(b * NTOK + st)) * QKVD + 2 * DDOWN + h * HD;
  int arow = tm + r, bcol = tn + r;
  v8f acc = {};
  for (int k0 = 0; k0 < cs; k0 += 32) {
    v16h af = {}, bf = {};
    if (FULL) af = frag_a_full(Pp + (size_t)arow * CHUNK + k0, hf);
    else if (arow < cs) af = frag_a(Pp + (size_t)arow * CHUNK, k0, cs, hf);
    if (bcol < HD) bf = frag_b_strided(Vb + bcol, QKVD, k0, cs, hf);
    acc = wmma16(af, bf, acc);
  }
  if (bcol < HD) {
#pragma unroll
    for (int v = 0; v < 8; ++v) {
      int row = tm + hf * 8 + v;
      if (FULL || row < cs)
        O[(size_t)(b * NTOK + st + row) * DDP + h * HD + bcol] = (_Float16)acc[v];
    }
  }
}

// ---------------- softmax over a chunk row ----------------
template<bool FULL>
__global__ void kSoftmax(const float* __restrict__ S, _Float16* __restrict__ P, int cs_rt) {
  const int cs = FULL ? CHUNK : cs_rt;
  int row = blockIdx.x, bz = blockIdx.y, tid = threadIdx.x;
  const float* sp = S + ((size_t)bz * CHUNK + row) * CHUNK;
  _Float16* pp = P + ((size_t)bz * CHUNK + row) * CHUNK;
  __shared__ float red[128];
  float m = -3.4e38f;
  for (int i = tid; i < cs; i += 128) m = fmaxf(m, sp[i]);
  red[tid] = m; __syncthreads();
  for (int s = 64; s > 0; s >>= 1) { if (tid < s) red[tid] = fmaxf(red[tid], red[tid + s]); __syncthreads(); }
  m = red[0]; __syncthreads();
  float sum = 0.f;
  for (int i = tid; i < cs; i += 128) sum += __expf(sp[i] - m);
  red[tid] = sum; __syncthreads();
  for (int s = 64; s > 0; s >>= 1) { if (tid < s) red[tid] += red[tid + s]; __syncthreads(); }
  float inv = 1.f / red[0];
  for (int i = tid; i < cs; i += 128) pp[i] = (_Float16)(__expf(sp[i] - m) * inv);
}

// ---------------- channel attention ----------------
__global__ void kReduceCA(const float* __restrict__ x, const float* __restrict__ olm,
                          float* __restrict__ mx, float* __restrict__ avg) {
  int bc = blockIdx.x, tid = threadIdx.x;
  const float* xp = x + (size_t)bc * HWSZ;
  __shared__ float sm[256], ss[256], sc[256];
  float m = -3.4e38f, s = 0.f, c = 0.f;
  for (int i = tid; i < HWSZ; i += 256)
    if (olm[i] > 0.f) { float v = xp[i]; m = fmaxf(m, v); s += v; c += 1.f; }
  sm[tid] = m; ss[tid] = s; sc[tid] = c; __syncthreads();
  for (int st = 128; st > 0; st >>= 1) {
    if (tid < st) { sm[tid] = fmaxf(sm[tid], sm[tid + st]); ss[tid] += ss[tid + st]; sc[tid] += sc[tid + st]; }
    __syncthreads();
  }
  if (tid == 0) { mx[bc] = sm[0]; avg[bc] = ss[0] / fmaxf(sc[0], 1.f); }
}

__global__ void kCAMLP(const float* __restrict__ mx, const float* __restrict__ avg,
                       const float* __restrict__ w1, const float* __restrict__ b1,
                       const float* __restrict__ w2, const float* __restrict__ b2,
                       float* __restrict__ att) {
  int b = blockIdx.x, tid = threadIdx.x;
  __shared__ float ta[CCH], tm[CCH], hs[CCH / 2];
  for (int i = tid; i < CCH; i += blockDim.x) { ta[i] = avg[b * CCH + i]; tm[i] = mx[b * CCH + i]; }
  __syncthreads();
  for (int j = tid; j < CCH / 2; j += blockDim.x) {
    float ha = b1[j], hm = b1[j];
    const float* wr = w1 + (size_t)j * CCH;
    for (int k = 0; k < CCH; ++k) { float w = wr[k]; ha += w * ta[k]; hm += w * tm[k]; }
    ha = ha >= 0.f ? ha : NEG_SLOPE * ha;
    hm = hm >= 0.f ? hm : NEG_SLOPE * hm;
    hs[j] = ha + hm;
  }
  __syncthreads();
  for (int c = tid; c < CCH; c += blockDim.x) {
    float o = 2.f * b2[c];
    const float* wr = w2 + (size_t)c * (CCH / 2);
    for (int j = 0; j < CCH / 2; ++j) o += wr[j] * hs[j];
    att[b * CCH + c] = 1.f / (1.f + __expf(-o));
  }
}

__global__ void kCAScale(const float* __restrict__ in, const float* __restrict__ att,
                         float* __restrict__ out, long n) {
  long i = (long)blockIdx.x * 256 + threadIdx.x;
  if (i < n) out[i] = att[i / HWSZ] * in[i];
}

// ---------------- spatial attention ----------------
__global__ void kSAStats(const float* __restrict__ in, float* __restrict__ st) {
  long i = (long)blockIdx.x * 256 + threadIdx.x;
  if (i >= (long)BATCH * HWSZ) return;
  int b = (int)(i / HWSZ), p = (int)(i % HWSZ);
  const float* xp = in + (size_t)b * CCH * HWSZ + p;
  float m = -3.4e38f, s = 0.f;
  for (int c = 0; c < CCH; ++c) { float v = xp[(size_t)c * HWSZ]; m = fmaxf(m, v); s += v; }
  st[((size_t)b * 2) * HWSZ + p]     = m;
  st[((size_t)b * 2 + 1) * HWSZ + p] = s * (1.f / CCH);
}

__global__ void kSAConv(const float* __restrict__ st, const float* __restrict__ w,
                        const float* __restrict__ bptr, const float* __restrict__ olm,
                        float* __restrict__ map) {
  long i = (long)blockIdx.x * 256 + threadIdx.x;
  if (i >= (long)BATCH * HWSZ) return;
  int b = (int)(i / HWSZ), p = (int)(i % HWSZ);
  int y = p / WW, x = p % WW;
  float acc = bptr[0];
  for (int ic = 0; ic < 2; ++ic)
    for (int ky = 0; ky < 7; ++ky) {
      int yy = y + ky - 3; if (yy < 0 || yy >= HH) continue;
      for (int kx = 0; kx < 7; ++kx) {
        int xx = x + kx - 3; if (xx < 0 || xx >= WW) continue;
        acc += st[((size_t)b * 2 + ic) * HWSZ + yy * WW + xx] * w[ic * 49 + ky * 7 + kx];
      }
    }
  map[i] = 1.f / (1.f + __expf(-(acc * olm[p])));
}

__global__ void kMulMap(const float* __restrict__ in, const float* __restrict__ map,
                        float* __restrict__ out, long n) {
  long i = (long)blockIdx.x * 256 + threadIdx.x;
  if (i < n) {
    long b = i / ((long)CCH * HWSZ);
    long p = i % HWSZ;
    out[i] = in[i] * map[b * HWSZ + p];
  }
}

// ---------------- unfold + register tokens ----------------
__global__ void kBuildXr(const float* __restrict__ x1, const float* __restrict__ reg,
                         float* __restrict__ xr) {
  long i = (long)blockIdx.x * 256 + threadIdx.x;
  if (i >= (long)MROWS * DIMT) return;
  long row = i / DIMT; int d = (int)(i % DIMT);
  int b = (int)(row / NTOK), t = (int)(row % NTOK);
  float v;
  if (t >= 1 && t <= NREG) v = reg[(t - 1) * DIMT + d];
  else {
    int p = (t == 0) ? 0 : (t - NREG);
    int ph = p >> 6, pw = p & 63;
    int c = d / 9, rem = d % 9, ii = rem / 3, jj = rem % 3;
    v = x1[(((size_t)b * CCH + c) * HH + ph + ii) * WW + pw + jj];
  }
  xr[i] = v;
}

// ---------------- layernorm: fp32(xr, stride DIMT) -> f16(xln, stride DIMP) ----
__global__ void kLN(const float* __restrict__ xr, const float* __restrict__ g,
                    const float* __restrict__ bb, _Float16* __restrict__ out) {
  int row = blockIdx.x, tid = threadIdx.x;
  const float* xp = xr + (size_t)row * DIMT;
  __shared__ float s1[256], s2[256];
  __shared__ float mean, rstd;
  float a = 0.f, q = 0.f;
  for (int i = tid; i < DIMT; i += 256) { float v = xp[i]; a += v; q += v * v; }
  s1[tid] = a; s2[tid] = q; __syncthreads();
  for (int st = 128; st > 0; st >>= 1) {
    if (tid < st) { s1[tid] += s1[tid + st]; s2[tid] += s2[tid + st]; }
    __syncthreads();
  }
  if (tid == 0) {
    float m = s1[0] / DIMT;
    float var = s2[0] / DIMT - m * m;
    mean = m; rstd = rsqrtf(var + 1e-5f);
  }
  __syncthreads();
  _Float16* op = out + (size_t)row * DIMP;
  for (int i = tid; i < DIMP; i += 256) {
    float val = (i < DIMT) ? ((xp[i] - mean) * rstd * g[i] + bb[i]) : 0.f;
    op[i] = (_Float16)val;
  }
}

// ---------------- fold (overlap-add, drop register tokens) ----------------
__global__ void kFold(const float* __restrict__ xr, float* __restrict__ res) {
  long i = (long)blockIdx.x * 256 + threadIdx.x;
  if (i >= (long)BATCH * CCH * HWSZ) return;
  int b = (int)(i / ((long)CCH * HWSZ));
  int rem = (int)(i % ((long)CCH * HWSZ));
  int c = rem / HWSZ, p = rem % HWSZ;
  int y = p / WW, x = p % WW;
  float acc = 0.f;
  for (int ii = 0; ii < 3; ++ii) {
    int ph = y - ii; if (ph < 0 || ph >= 64) continue;
    for (int jj = 0; jj < 3; ++jj) {
      int pw = x - jj; if (pw < 0 || pw >= 64) continue;
      int pt = ph * 64 + pw;
      int row = (pt == 0) ? 0 : pt + NREG;
      acc += xr[((size_t)(b * NTOK + row)) * DIMT + c * 9 + ii * 3 + jj];
    }
  }
  res[i] = acc;
}

// ---------------- fp32 -> f16 weight pad-conversion, zero fill ----------------
__global__ void kCvtPad(const float* __restrict__ in, _Float16* __restrict__ out,
                        int N, int K, int Kpad, long total) {
  long i = (long)blockIdx.x * 256 + threadIdx.x;
  if (i >= total) return;
  int n = (int)(i / Kpad), k = (int)(i % Kpad);
  out[i] = (n < N && k < K) ? (_Float16)in[(size_t)n * K + k] : (_Float16)0.f;
}
__global__ void kFillH(_Float16* __restrict__ p, long n) {
  long i = (long)blockIdx.x * 256 + threadIdx.x;
  if (i < n) p[i] = (_Float16)0.f;
}

// ============================================================
extern "C" void kernel_launch(void* const* d_in, const int* in_sizes, int n_in,
                              void* d_out, int out_size, void* d_ws, size_t ws_size,
                              hipStream_t stream) {
  const float* xin    = (const float*)d_in[0];
  const float* olm    = (const float*)d_in[1];
  const float* ca_w1  = (const float*)d_in[2];
  const float* ca_b1  = (const float*)d_in[3];
  const float* ca_w2  = (const float*)d_in[4];
  const float* ca_b2  = (const float*)d_in[5];
  const float* sa_w   = (const float*)d_in[6];
  const float* sa_b   = (const float*)d_in[7];
  const float* reg    = (const float*)d_in[8];
  const float* ln1_g  = (const float*)d_in[9];
  const float* ln1_b  = (const float*)d_in[10];
  const float* wdown  = (const float*)d_in[11];
  const float* wqkv   = (const float*)d_in[12];
  const float* wup    = (const float*)d_in[13];
  const float* bup    = (const float*)d_in[14];
  const float* ln2_g  = (const float*)d_in[15];
  const float* ln2_b  = (const float*)d_in[16];
  const float* m_w1   = (const float*)d_in[17];
  const float* m_b1   = (const float*)d_in[18];
  const float* m_w2   = (const float*)d_in[19];
  const float* m_b2   = (const float*)d_in[20];

  char* base = (char*)d_ws;
  size_t off = 0;
  auto alloc = [&](size_t bytes) -> void* {
    void* p = base + off;
    off += (bytes + 255) & ~(size_t)255;
    return p;
  };
  float*     mxB    = (float*)alloc((size_t)BATCH * CCH * 4);
  float*     avgB   = (float*)alloc((size_t)BATCH * CCH * 4);
  float*     attB   = (float*)alloc((size_t)BATCH * CCH * 4);
  float*     sstats = (float*)alloc((size_t)BATCH * 2 * HWSZ * 4);
  float*     samap  = (float*)alloc((size_t)BATCH * HWSZ * 4);
  float*     x1     = (float*)alloc((size_t)BATCH * CCH * HWSZ * 4);
  float*     res    = (float*)alloc((size_t)BATCH * CCH * HWSZ * 4);
  float*     xr     = (float*)alloc((size_t)MROWS * DIMT * 4);
  _Float16*  xln    = (_Float16*)alloc((size_t)MROWSP * DIMP * 2);
  _Float16*  xd     = (_Float16*)alloc((size_t)MROWSP * DDP * 2);
  _Float16*  qkvb   = (_Float16*)alloc((size_t)MROWS * QKVD * 2);
  _Float16*  attnO  = (_Float16*)alloc((size_t)MROWSP * DDP * 2);
  float*     scoresS= (float*)alloc((size_t)BATCH * HEADS * CHUNK * CHUNK * 4);
  _Float16*  Pb     = (_Float16*)alloc((size_t)BATCH * HEADS * CHUNK * CHUNK * 2);
  _Float16*  h1     = (_Float16*)alloc((size_t)MROWSP * H1P * 2);
  _Float16*  wdownH = (_Float16*)alloc((size_t)NP_DOWN * DIMP * 2);
  _Float16*  wqkvH  = (_Float16*)alloc((size_t)NP_QKV * DDP * 2);
  _Float16*  wupH   = (_Float16*)alloc((size_t)NP_UP * DDP * 2);
  _Float16*  w1H    = (_Float16*)alloc((size_t)NP_MLP1 * DIMP * 2);
  _Float16*  w2H    = (_Float16*)alloc((size_t)NP_MLP2 * H1P * 2);

  const long nEl  = (long)BATCH * CCH * HWSZ;
  const long nPix = (long)BATCH * HWSZ;
  const long nXr  = (long)MROWS * DIMT;
  auto G = [](long n) { return dim3((unsigned)((n + 255) / 256)); };

  // ---- padded f16 weights (zero-pad N and K) ----
  { long t = (long)NP_DOWN * DIMP; kCvtPad<<<G(t),256,0,stream>>>(wdown, wdownH, DDOWN, DIMT, DIMP, t); }
  { long t = (long)NP_QKV  * DDP;  kCvtPad<<<G(t),256,0,stream>>>(wqkv,  wqkvH,  QKVD,  DDOWN, DDP, t); }
  { long t = (long)NP_UP   * DDP;  kCvtPad<<<G(t),256,0,stream>>>(wup,   wupH,   DIMT,  DDOWN, DDP, t); }
  { long t = (long)NP_MLP1 * DIMP; kCvtPad<<<G(t),256,0,stream>>>(m_w1,  w1H,    H1D,   DIMT, DIMP, t); }
  { long t = (long)NP_MLP2 * H1P;  kCvtPad<<<G(t),256,0,stream>>>(m_w2,  w2H,    DIMT,  H1D,  H1P, t); }
  // ---- zero-fill padded activation buffers (pad rows/cols must read as 0) ----
  kFillH<<<G((long)MROWSP * DDP), 256, 0, stream>>>(xd,    (long)MROWSP * DDP);
  kFillH<<<G((long)MROWSP * DDP), 256, 0, stream>>>(attnO, (long)MROWSP * DDP);
  kFillH<<<G((long)MROWSP * H1P), 256, 0, stream>>>(h1,    (long)MROWSP * H1P);
  kFillH<<<G((long)(MROWSP - MROWS) * DIMP), 256, 0, stream>>>(xln + (size_t)MROWS * DIMP,
                                                               (long)(MROWSP - MROWS) * DIMP);

  // ---- stage 1: CA then SA on x ----
  kReduceCA<<<BATCH * CCH, 256, 0, stream>>>(xin, olm, mxB, avgB);
  kCAMLP<<<BATCH, 256, 0, stream>>>(mxB, avgB, ca_w1, ca_b1, ca_w2, ca_b2, attB);
  kCAScale<<<G(nEl), 256, 0, stream>>>(xin, attB, x1, nEl);
  kSAStats<<<G(nPix), 256, 0, stream>>>(x1, sstats);
  kSAConv<<<G(nPix), 256, 0, stream>>>(sstats, sa_w, sa_b, olm, samap);
  kMulMap<<<G(nEl), 256, 0, stream>>>(x1, samap, x1, nEl);

  // ---- unfold + register tokens, LN1 ----
  kBuildXr<<<G(nXr), 256, 0, stream>>>(x1, reg, xr);
  kLN<<<MROWS, 256, 0, stream>>>(xr, ln1_g, ln1_b, xln);

  // ---- attention GEMM chain (WMMA f16, f32 accum, branch-free inner loops) ----
  const unsigned MT = MROWSP / 16;   // 1025
  dim3 g1((NP_DOWN + 255) / 256, MT);
  kGemm<0, false, false><<<g1, 128, 0, stream>>>(xln, DIMP, wdownH, DIMP, nullptr,
                                                 xd, nullptr, DDP, MROWS, DDOWN, NP_DOWN, DIMP, 1.f);
  dim3 g2((NP_QKV + 255) / 256, MT);
  kGemm<0, false, false><<<g2, 128, 0, stream>>>(xd, DDP, wqkvH, DDP, nullptr,
                                                 qkvb, nullptr, QKVD, MROWS, QKVD, NP_QKV, DDP, 1.f);
  for (int st = 0; st < NTOK; st += CHUNK) {
    int cs = NTOK - st; if (cs > CHUNK) cs = CHUNK;
    int nt = (cs + 15) / 16;
    dim3 gs(nt, nt, BATCH * HEADS);
    dim3 gsm(cs, BATCH * HEADS);
    dim3 ga((HD + 15) / 16, nt, BATCH * HEADS);
    if (cs == CHUNK) {
      kAttnScores<true><<<gs, 32, 0, stream>>>(qkvb, scoresS, st, cs);
      kSoftmax<true><<<gsm, 128, 0, stream>>>(scoresS, Pb, cs);
      kAttnAV<true><<<ga, 32, 0, stream>>>(Pb, qkvb, attnO, st, cs);
    } else {
      kAttnScores<false><<<gs, 32, 0, stream>>>(qkvb, scoresS, st, cs);
      kSoftmax<false><<<gsm, 128, 0, stream>>>(scoresS, Pb, cs);
      kAttnAV<false><<<ga, 32, 0, stream>>>(Pb, qkvb, attnO, st, cs);
    }
  }
  dim3 g3((NP_UP + 255) / 256, MT);
  kGemm<2, false, true><<<g3, 128, 0, stream>>>(attnO, DDP, wupH, DDP, bup,
                                                nullptr, xr, DIMT, MROWS, DIMT, NP_UP, DDP, 1.f);

  // ---- MLP ----
  kLN<<<MROWS, 256, 0, stream>>>(xr, ln2_g, ln2_b, xln);
  dim3 g4((NP_MLP1 + 255) / 256, MT);
  kGemm<0, true, true><<<g4, 128, 0, stream>>>(xln, DIMP, w1H, DIMP, m_b1,
                                               h1, nullptr, H1P, MROWS, H1D, NP_MLP1, DIMP, 1.f);
  dim3 g5((NP_MLP2 + 255) / 256, MT);
  kGemm<2, false, true><<<g5, 128, 0, stream>>>(h1, H1P, w2H, H1P, m_b2,
                                                nullptr, xr, DIMT, MROWS, DIMT, NP_MLP2, H1P, 1.f);

  // ---- fold, then CA/SA stage 2 ----
  kFold<<<G(nEl), 256, 0, stream>>>(xr, res);
  kReduceCA<<<BATCH * CCH, 256, 0, stream>>>(res, olm, mxB, avgB);
  kCAMLP<<<BATCH, 256, 0, stream>>>(mxB, avgB, ca_w1, ca_b1, ca_w2, ca_b2, attB);
  kCAScale<<<G(nEl), 256, 0, stream>>>(res, attB, x1, nEl);
  kSAStats<<<G(nPix), 256, 0, stream>>>(x1, sstats);
  kSAConv<<<G(nPix), 256, 0, stream>>>(sstats, sa_w, sa_b, olm, samap);
  kMulMap<<<G(nEl), 256, 0, stream>>>(x1, samap, (float*)d_out, nEl);
}
